// DownsampleLayer_55198919688305
// MI455X (gfx1250) — matrix operand order repaired and checked by reading, not compiled, not run on hardware
//
#include <hip/hip_runtime.h>
#include <math.h>
#include <stdint.h>

// ---------------- problem constants ----------------
#define BB   4
#define NN   8192
#define CDIM 128          // DIM == HID == 128
#define NGRP 8
#define KNb  16           // K neighbors
#define MM   4096         // M = N * 0.5
#define MKc  (MM * KNb)   // 65536
#define EPSN 1e-5f

#ifndef USE_ASYNC_LDS
#define USE_ASYNC_LDS 1   // CDNA5 global_load_async_to_lds_b128 staging of W panel
#endif

typedef _Float16 half16 __attribute__((ext_vector_type(16)));
typedef _Float16 half8  __attribute__((ext_vector_type(8)));
typedef float    float8 __attribute__((ext_vector_type(8)));

// ======================================================================
// 1. Furthest point sampling: one block per batch, 8 pts/thread in regs
// ======================================================================
__global__ __launch_bounds__(1024) void fps_kernel(const float* __restrict__ xyzs,
                                                   int* __restrict__ sidx) {
  const int b   = blockIdx.x;
  const int tid = threadIdx.x;
  const float* px = xyzs + (long)b * 3 * NN;
  const float* py = px + NN;
  const float* pz = px + 2 * NN;

  float x8[8], y8[8], z8[8], d8[8];
  int   n8[8];
#pragma unroll
  for (int j = 0; j < 8; ++j) {
    int n = j * 1024 + tid;
    n8[j] = n; x8[j] = px[n]; y8[j] = py[n]; z8[j] = pz[n];
    d8[j] = 3.4e38f;
  }

  __shared__ float s_bd[32];
  __shared__ int   s_bi[32];
  __shared__ int   s_far;

  int far = 0;
  for (int i = 0; i < MM; ++i) {
    if (tid == 0) sidx[(long)b * MM + i] = far;
    float fx = px[far], fy = py[far], fz = pz[far];

    float bd = -1.0f; int bi = 0x7fffffff;
#pragma unroll
    for (int j = 0; j < 8; ++j) {
      float dx = x8[j] - fx, dy = y8[j] - fy, dz = z8[j] - fz;
      float d  = dx * dx + dy * dy + dz * dz;
      d8[j] = fminf(d8[j], d);
      if (d8[j] > bd || (d8[j] == bd && n8[j] < bi)) { bd = d8[j]; bi = n8[j]; }
    }
    for (int off = 16; off; off >>= 1) {
      float od = __shfl_xor(bd, off, 32);
      int   oi = __shfl_xor(bi, off, 32);
      if (od > bd || (od == bd && oi < bi)) { bd = od; bi = oi; }
    }
    if ((tid & 31) == 0) { s_bd[tid >> 5] = bd; s_bi[tid >> 5] = bi; }
    __syncthreads();
    if (tid < 32) {
      bd = s_bd[tid]; bi = s_bi[tid];
      for (int off = 16; off; off >>= 1) {
        float od = __shfl_xor(bd, off, 32);
        int   oi = __shfl_xor(bi, off, 32);
        if (od > bd || (od == bd && oi < bi)) { bd = od; bi = oi; }
      }
      if (tid == 0) s_far = bi;
    }
    __syncthreads();
    far = s_far;
  }
}

// ======================================================================
// 2. gather sampled xyz -> out0 [B,3,M] and spts [B,M,3]
// ======================================================================
__global__ void sample_xyz_kernel(const float* __restrict__ xyzs,
                                  const int* __restrict__ sidx,
                                  float* __restrict__ out0,
                                  float* __restrict__ spts) {
  long t = (long)blockIdx.x * blockDim.x + threadIdx.x;
  if (t >= (long)BB * MM) return;
  int b = (int)(t / MM), m = (int)(t % MM);
  int idx = sidx[t];
  const float* base = xyzs + (long)b * 3 * NN;
  float x = base[idx], y = base[NN + idx], z = base[2 * NN + idx];
  float* o = out0 + (long)b * 3 * MM;
  o[m] = x; o[MM + m] = y; o[2 * MM + m] = z;
  float* s = spts + t * 3;
  s[0] = x; s[1] = y; s[2] = z;
}

// ======================================================================
// 3. nearest sampled point per original point + density count (atomic)
// ======================================================================
__global__ __launch_bounds__(256) void nearest_kernel(const float* __restrict__ spts,
                                                      const float* __restrict__ xyzs,
                                                      int* __restrict__ o2s,
                                                      float* __restrict__ dsn) {
  const int b = blockIdx.y;
  const int n = blockIdx.x * 256 + threadIdx.x;
  const float* base = xyzs + (long)b * 3 * NN;
  float ox = base[n], oy = base[NN + n], oz = base[2 * NN + n];

  __shared__ float sp[1024 * 3];
  float bd = 3.4e38f; int bm = 0;
  for (int m0 = 0; m0 < MM; m0 += 1024) {
    for (int i = threadIdx.x; i < 1024 * 3; i += 256)
      sp[i] = spts[((long)b * MM + m0) * 3 + i];
    __syncthreads();
    for (int mm = 0; mm < 1024; ++mm) {
      float dx = ox - sp[mm * 3], dy = oy - sp[mm * 3 + 1], dz = oz - sp[mm * 3 + 2];
      float d = dx * dx + dy * dy + dz * dz;
      if (d < bd) { bd = d; bm = m0 + mm; }
    }
    __syncthreads();
  }
  o2s[(long)b * NN + n] = bm;
  atomicAdd(&dsn[(long)b * MM + bm], 1.0f);
}

// ======================================================================
// 4. KNN: one block per (m,b); all 8192 dists in LDS, 16 extraction rounds
// ======================================================================
__global__ __launch_bounds__(256) void knn_kernel(const float* __restrict__ spts,
                                                  const float* __restrict__ xyzs,
                                                  int* __restrict__ knn) {
  const int m = blockIdx.x, b = blockIdx.y, tid = threadIdx.x;
  __shared__ float d[NN];
  __shared__ float s_bd[8];
  __shared__ int   s_bi[8];

  const float* sp = spts + ((long)b * MM + m) * 3;
  float qx = sp[0], qy = sp[1], qz = sp[2];
  const float* base = xyzs + (long)b * 3 * NN;
  for (int n = tid; n < NN; n += 256) {
    float dx = base[n] - qx, dy = base[NN + n] - qy, dz = base[2 * NN + n] - qz;
    d[n] = dx * dx + dy * dy + dz * dz;
  }
  __syncthreads();

  for (int kk = 0; kk < KNb; ++kk) {
    float bd = 3.4e38f; int bi = NN;
    for (int n = tid; n < NN; n += 256) {
      float dv = d[n];
      if (dv < bd || (dv == bd && n < bi)) { bd = dv; bi = n; }
    }
    for (int off = 16; off; off >>= 1) {
      float od = __shfl_xor(bd, off, 32);
      int   oi = __shfl_xor(bi, off, 32);
      if (od < bd || (od == bd && oi < bi)) { bd = od; bi = oi; }
    }
    if ((tid & 31) == 0) { s_bd[tid >> 5] = bd; s_bi[tid >> 5] = bi; }
    __syncthreads();
    if (tid == 0) {
      for (int i = 1; i < 8; ++i)
        if (s_bd[i] < bd || (s_bd[i] == bd && s_bi[i] < bi)) { bd = s_bd[i]; bi = s_bi[i]; }
      knn[((long)b * MM + m) * KNb + kk] = bi;
      d[bi] = 3.4e38f;
    }
    __syncthreads();
  }
}

// ======================================================================
// 5. geometry: diff, direction, dist, mask, mean-distance accumulation
// ======================================================================
__global__ void geom_kernel(const float* __restrict__ xyzs,
                            const float* __restrict__ spts,
                            const int* __restrict__ knn,
                            const int* __restrict__ o2s,
                            float* __restrict__ negdiff,
                            float* __restrict__ localf,
                            int* __restrict__ maskb,
                            float* __restrict__ mdsum) {
  long t = (long)blockIdx.x * blockDim.x + threadIdx.x;
  if (t >= (long)BB * MKc) return;
  int b = (int)(t / MKc);
  long p = t % MKc;
  int m = (int)(p >> 4);
  int idx = knn[t];
  const float* base = xyzs + (long)b * 3 * NN;
  float kx = base[idx], ky = base[NN + idx], kz = base[2 * NN + idx];
  const float* sp = spts + ((long)b * MM + m) * 3;
  float dx = kx - sp[0], dy = ky - sp[1], dz = kz - sp[2];
  float dist = sqrtf(dx * dx + dy * dy + dz * dz);
  float inv = 1.0f / fmaxf(dist, 1e-12f);
  long base3 = (long)b * 3 * MKc + p;
  negdiff[base3]            = -dx;
  negdiff[base3 + MKc]      = -dy;
  negdiff[base3 + 2 * MKc]  = -dz;
  long base4 = (long)b * 4 * MKc + p;
  localf[base4]             = dx * inv;
  localf[base4 + MKc]       = dy * inv;
  localf[base4 + 2 * MKc]   = dz * inv;
  localf[base4 + 3 * MKc]   = dist;
  int msk = (o2s[(long)b * NN + idx] == m) ? 1 : 0;
  maskb[t] = msk;
  if (msk) atomicAdd(&mdsum[(long)b * MM + m], dist);
}

__global__ void meandiv_kernel(float* __restrict__ out3, const float* __restrict__ dsn) {
  long t = (long)blockIdx.x * blockDim.x + threadIdx.x;
  if (t >= (long)BB * MM) return;
  out3[t] = out3[t] / dsn[t];
}

// ======================================================================
// 6. tiled transpose + f32->f16 convert:  X [C][P] -> Yt [P][ytStride]+ytOff
// ======================================================================
__global__ __launch_bounds__(256) void transpose_f16_kernel(
    const float* __restrict__ X, _Float16* __restrict__ Yt,
    int C, int P, int ytStride, int ytOff, long Xbs, long Ybs) {
  const int b = blockIdx.z;
  X  += (long)b * Xbs;
  Yt += (long)b * Ybs;
  const int p0 = blockIdx.x * 32, c0 = blockIdx.y * 32;
  __shared__ _Float16 tile[32 * 33];
  const int t = threadIdx.x;
#pragma unroll
  for (int i = 0; i < 4; ++i) {
    int idx = t + i * 256;
    int cc = idx >> 5, pp = idx & 31;
    tile[cc * 33 + pp] = (_Float16)X[(long)(c0 + cc) * P + p0 + pp];
  }
  __syncthreads();
#pragma unroll
  for (int i = 0; i < 4; ++i) {
    int idx = t + i * 256;
    int pp = idx >> 5, cc = idx & 31;
    Yt[(long)(p0 + pp) * ytStride + ytOff + c0 + cc] = tile[cc * 33 + pp];
  }
}

// ======================================================================
// 7. WMMA GEMM:  Y[128][P] = W[128][Cin] * Xt^T + bias (+add)
//    Xt is f16, row-major [P][Cin]  (K-contiguous -> B frag = 2 x b128)
//    optional transposed f16 output Yt[P][ytStride]+ytOff (one h8 store/lane)
//    W panel staged to LDS via CDNA5 async-to-LDS (ASYNCcnt) when enabled
// ======================================================================
__global__ __launch_bounds__(256) void gemm128_wmma(
    float* __restrict__ Y, _Float16* __restrict__ Yt,
    const _Float16* __restrict__ Xt,
    const float* __restrict__ W, const float* __restrict__ bias,
    const float* __restrict__ addsrc,
    int Cin, int P, int ytStride, int ytOff,
    long Xbs, long Ybs, long Ytbs, long Abs) {
  __shared__ float ldsW[128 * 32];   // f32 W panel (16KB)
  const int b = blockIdx.z;
  Xt += (long)b * Xbs;
  Y  += (long)b * Ybs;
  if (Yt)     Yt     += (long)b * Ytbs;
  if (addsrc) addsrc += (long)b * Abs;

  const int tid  = threadIdx.x;
  const int wave = tid >> 5;
  const int lane = tid & 31;
  const int r    = lane & 15;
  const int hi   = lane >> 4;
  const int pblk = blockIdx.x * 128;

  float8 acc[8];
#pragma unroll
  for (int i = 0; i < 8; ++i) acc[i] = (float8)0.0f;

  for (int k0 = 0; k0 < Cin; k0 += 32) {
    if (k0) __syncthreads();
#if USE_ASYNC_LDS
    {
      // thread t copies 64B of one W row segment: 4 x async b128,
      // INST_OFFSET applies to both LDS and global addresses (ISA 08 §4.4)
      const int row  = tid >> 1;
      const int halfseg = (tid & 1) * 16;
      const float* gp = W + (long)row * Cin + k0 + halfseg;
      unsigned la = (unsigned)(uintptr_t)&ldsW[row * 32 + halfseg];
      asm volatile(
          "global_load_async_to_lds_b128 %0, %1, off\n\t"
          "global_load_async_to_lds_b128 %0, %1, off offset:16\n\t"
          "global_load_async_to_lds_b128 %0, %1, off offset:32\n\t"
          "global_load_async_to_lds_b128 %0, %1, off offset:48"
          :: "v"(la), "v"(gp) : "memory");
      asm volatile("s_wait_asynccnt 0x0" ::: "memory");
    }
#else
    for (int i = tid; i < 128 * 32; i += 256) {
      int row = i >> 5, col = i & 31;
      ldsW[i] = W[(long)row * Cin + k0 + col];
    }
#endif
    __syncthreads();

    // A fragment: rows [wave*16, wave*16+16), K panel [k0, k0+32)
    half16 a;
    const int arow = wave * 16 + r;
#pragma unroll
    for (int j = 0; j < 16; ++j) {
      int kk = hi * 16 + j;
      a[j] = (_Float16)ldsW[arow * 32 + kk];
    }

#pragma unroll
    for (int pt = 0; pt < 8; ++pt) {
      int p = pblk + pt * 16 + r;
      // 16 K-contiguous halfs for this lane's column: one 32B vector load
      half16 bf = *(const half16*)(Xt + (long)p * Cin + k0 + hi * 16);
      acc[pt] = __builtin_amdgcn_wmma_f32_16x16x32_f16(
          false, a, false, bf, (short)0, acc[pt], false, false);
    }
    if (k0 + 32 < Cin)
      __builtin_prefetch(Xt + (long)(pblk + r) * Cin + k0 + 32, 0, 1);
  }

#pragma unroll
  for (int pt = 0; pt < 8; ++pt) {
    int p = pblk + pt * 16 + r;
    half8 hv;
#pragma unroll
    for (int g = 0; g < 8; ++g) {
      int row = wave * 16 + hi * 8 + g;
      float v = acc[pt][g] + bias[row];
      if (addsrc) v += addsrc[(long)row * P + p];
      Y[(long)row * P + p] = v;
      hv[g] = (_Float16)v;
    }
    if (Yt)
      *(half8*)(Yt + (long)p * ytStride + ytOff + wave * 16 + hi * 8) = hv;
  }
}

// ======================================================================
// 8. small-Cin conv (Cin = 1/3/4):  Y[128][P]
// ======================================================================
__global__ void conv_small_cin(float* __restrict__ Y, const float* __restrict__ X,
                               const float* __restrict__ W, const float* __restrict__ bias,
                               int Cin, int P) {
  long t = (long)blockIdx.x * blockDim.x + threadIdx.x;
  if (t >= (long)CDIM * P) return;
  int o = (int)(t / P);
  long p = t % P;
  float s = bias[o];
  for (int i = 0; i < Cin; ++i) s += W[o * Cin + i] * X[(long)i * P + p];
  Y[t] = s;
}

// ======================================================================
// 9. GroupNorm (per-batch global): reduce + apply(+ReLU)
// ======================================================================
__global__ void zero_f32(float* __restrict__ p, long n) {
  long t = (long)blockIdx.x * blockDim.x + threadIdx.x;
  if (t < n) p[t] = 0.0f;
}

__global__ __launch_bounds__(256) void gn_reduce(const float* __restrict__ X,
                                                 float* __restrict__ stats, int P) {
  const int g = blockIdx.y;
  float s = 0.0f, s2 = 0.0f;
  long start  = (long)blockIdx.x * blockDim.x + threadIdx.x;
  long stride = (long)gridDim.x * blockDim.x;
  for (int c = 0; c < CDIM / NGRP; ++c) {
    const float* row = X + ((long)g * (CDIM / NGRP) + c) * P;
    for (long p = start; p < P; p += stride) { float v = row[p]; s += v; s2 += v * v; }
  }
  for (int off = 16; off; off >>= 1) { s += __shfl_xor(s, off, 32); s2 += __shfl_xor(s2, off, 32); }
  __shared__ float ls[16];
  int wv = threadIdx.x >> 5, ln = threadIdx.x & 31;
  if (ln == 0) { ls[wv] = s; ls[8 + wv] = s2; }
  __syncthreads();
  if (threadIdx.x == 0) {
    for (int i = 1; i < 8; ++i) { s += ls[i]; s2 += ls[8 + i]; }
    atomicAdd(&stats[2 * g], s);
    atomicAdd(&stats[2 * g + 1], s2);
  }
}

__global__ void gn_apply_relu(float* __restrict__ X, const float* __restrict__ stats,
                              const float* __restrict__ gam, const float* __restrict__ bet,
                              int P) {
  long t = (long)blockIdx.x * blockDim.x + threadIdx.x;
  if (t >= (long)CDIM * P) return;
  int c = (int)(t / P);
  int g = c >> 4;
  float cnt = 16.0f * (float)P;
  float mu  = stats[2 * g] / cnt;
  float var = stats[2 * g + 1] / cnt - mu * mu;
  float xv  = (X[t] - mu) * rsqrtf(var + EPSN) * gam[c] + bet[c];
  X[t] = fmaxf(xv, 0.0f);
}

// ======================================================================
// 10. attention glue
// ======================================================================
__global__ void attn_in_kernel(float* __restrict__ T, const float* __restrict__ q,
                               const float* __restrict__ ksrc, const float* __restrict__ penc,
                               const int* __restrict__ knn) {
  long t = (long)blockIdx.x * blockDim.x + threadIdx.x;
  if (t >= (long)CDIM * MKc) return;
  int c = (int)(t / MKc);
  long p = t % MKc;
  int m = (int)(p >> 4);
  int idx = knn[p];
  T[t] = q[(long)c * MM + m] - ksrc[(long)c * NN + idx] + penc[t];
}

__global__ void softmax16_kernel(float* __restrict__ A, const int* __restrict__ maskb,
                                 float scale) {
  long t = (long)blockIdx.x * blockDim.x + threadIdx.x;
  if (t >= (long)CDIM * MM) return;
  int c = (int)(t / MM), m = (int)(t % MM);
  float* a = A + (long)c * MKc + (long)m * KNb;
  const int* mk = maskb + (long)m * KNb;
  float v[KNb];
  float mx = -3.402823466e38f;
#pragma unroll
  for (int k = 0; k < KNb; ++k) {
    float x = mk[k] ? a[k] * scale : -3.402823466e38f;
    v[k] = x;
    mx = fmaxf(mx, x);
  }
  float ssum = 0.0f;
#pragma unroll
  for (int k = 0; k < KNb; ++k) { v[k] = expf(v[k] - mx); ssum += v[k]; }
  float inv = 1.0f / ssum;
#pragma unroll
  for (int k = 0; k < KNb; ++k) a[k] = v[k] * inv;
}

__global__ void attn_out_kernel(float* __restrict__ R, const float* __restrict__ att,
                                const float* __restrict__ vsrc, const float* __restrict__ penc,
                                const int* __restrict__ knn) {
  long t = (long)blockIdx.x * blockDim.x + threadIdx.x;
  if (t >= (long)CDIM * MM) return;
  int c = (int)(t / MM), m = (int)(t % MM);
  const float* a  = att  + (long)c * MKc + (long)m * KNb;
  const float* pe = penc + (long)c * MKc + (long)m * KNb;
  const int*   kn = knn + (long)m * KNb;
  float s = 0.0f;
#pragma unroll
  for (int k = 0; k < KNb; ++k)
    s += a[k] * (vsrc[(long)c * NN + kn[k]] + pe[k]);
  R[t] = s;
}

__global__ void mulsum16_kernel(float* __restrict__ Y, const float* __restrict__ pe,
                                const float* __restrict__ pa) {
  long t = (long)blockIdx.x * blockDim.x + threadIdx.x;
  if (t >= (long)CDIM * MM) return;
  int c = (int)(t / MM), m = (int)(t % MM);
  const float* p1 = pe + (long)c * MKc + (long)m * KNb;
  const float* p2 = pa + (long)c * MKc + (long)m * KNb;
  float s = 0.0f;
#pragma unroll
  for (int k = 0; k < KNb; ++k) s += p1[k] * p2[k];
  Y[(long)c * MM + m] = s;
}

// ======================================================================
// 11. channel-major gather:  Y[b][c][m] = X[b][c][sidx[b][m]]
// ======================================================================
__global__ void gather_cm_kernel(float* __restrict__ Y, const float* __restrict__ X,
                                 const int* __restrict__ sidx) {
  long t = (long)blockIdx.x * blockDim.x + threadIdx.x;
  if (t >= (long)BB * CDIM * MM) return;
  long b = t / ((long)CDIM * MM);
  long r = t % ((long)CDIM * MM);
  int c = (int)(r / MM), m = (int)(r % MM);
  Y[t] = X[(long)b * CDIM * NN + (long)c * NN + sidx[b * MM + m]];
}

// ======================================================================
// host-side orchestration
// ======================================================================
static inline unsigned cdivu(long n, int b) { return (unsigned)((n + b - 1) / b); }

enum {
  P_pre_w = 0, P_pre_b, P_wq_w, P_wq_b, P_wk_w, P_wk_b, P_wv_w, P_wv_b,
  P_delta_w1, P_delta_b1, P_delta_g, P_delta_be, P_delta_w2, P_delta_b2,
  P_gamma_w1, P_gamma_b1, P_gamma_g, P_gamma_be, P_gamma_w2, P_gamma_b2,
  P_ptpost_w, P_ptpost_b,
  P_pospre_w1, P_pospre_b1, P_pospre_g, P_pospre_be, P_pospre_w2, P_pospre_b2,
  P_posattn_w1, P_posattn_b1, P_posattn_g, P_posattn_be, P_posattn_w2, P_posattn_b2,
  P_den_w1, P_den_b1, P_den_g, P_den_be, P_den_w2, P_den_b2, P_post_w, P_post_b,
  P_COUNT
};

extern "C" void kernel_launch(void* const* d_in, const int* in_sizes, int n_in,
                              void* d_out, int out_size, void* d_ws, size_t ws_size,
                              hipStream_t stream) {
  (void)in_sizes; (void)n_in; (void)out_size;

  const float* xyzs  = (const float*)d_in[0];
  const float* feats = (const float*)d_in[1];
  const float* prm[P_COUNT];
  for (int i = 0; i < P_COUNT; ++i) prm[i] = (const float*)d_in[2 + i];

  float* out0 = (float*)d_out;                       // sampled_xyzs  [B,3,M]
  float* out1 = out0 + (long)BB * 3 * MM;            // sampled_out   [B,128,M]
  float* out2 = out1 + (long)BB * CDIM * MM;         // downsample_num[B,M]
  float* out3 = out2 + (long)BB * MM;                // mean_distance [B,M]

  // ---- workspace bump allocator ----
  char* wp = (char*)d_ws;
  auto alloc = [&](size_t bytes) -> void* {
    void* r = (void*)wp;
    wp += (bytes + 255) & ~(size_t)255;
    return r;
  };
  int*   sidx    = (int*)  alloc((size_t)BB * MM * 4);
  int*   o2s     = (int*)  alloc((size_t)BB * NN * 4);
  int*   knn     = (int*)  alloc((size_t)BB * MKc * 4);
  int*   maskb   = (int*)  alloc((size_t)BB * MKc * 4);
  float* spts    = (float*)alloc((size_t)BB * MM * 3 * 4);
  float* negdiff = (float*)alloc((size_t)BB * 3 * MKc * 4);
  float* localf  = (float*)alloc((size_t)BB * 4 * MKc * 4);
  float* fpre    = (float*)alloc((size_t)BB * CDIM * NN * 4);
  float* wkf     = (float*)alloc((size_t)BB * CDIM * NN * 4);
  float* wvf     = (float*)alloc((size_t)BB * CDIM * NN * 4);
  float* sfeat   = (float*)alloc((size_t)BB * CDIM * MM * 4);
  float* query   = (float*)alloc((size_t)BB * CDIM * MM * 4);
  float* resb    = (float*)alloc((size_t)BB * CDIM * MM * 4);
  float* idg     = (float*)alloc((size_t)BB * CDIM * MM * 4);
  float* cat3    = (float*)alloc((size_t)3 * CDIM * MM * 4);   // per-batch reuse
  float* denh    = (float*)alloc((size_t)CDIM * MM * 4);       // per-batch reuse
  float* gnstat  = (float*)alloc((size_t)2 * NGRP * 4);
  float* big0    = (float*)alloc((size_t)CDIM * MKc * 4);      // per-batch f32 (32MB)
  float* big1    = (float*)alloc((size_t)CDIM * MKc * 4);
  float* big2    = (float*)alloc((size_t)CDIM * MKc * 4);
  // transposed f16 operands for WMMA GEMMs
  _Float16* featsT = (_Float16*)alloc((size_t)BB * NN * CDIM * 2);
  _Float16* fpreT  = (_Float16*)alloc((size_t)BB * NN * CDIM * 2);
  _Float16* sfeatT = (_Float16*)alloc((size_t)BB * MM * CDIM * 2);
  _Float16* bigT   = (_Float16*)alloc((size_t)MKc * CDIM * 2);  // per-batch reuse
  _Float16* bigTb  = (_Float16*)alloc((size_t)MKc * CDIM * 2);
  _Float16* resT   = (_Float16*)alloc((size_t)MM * CDIM * 2);
  _Float16* denhT  = (_Float16*)alloc((size_t)MM * CDIM * 2);
  _Float16* catT   = (_Float16*)alloc((size_t)MM * 3 * CDIM * 2);
  if ((size_t)(wp - (char*)d_ws) > ws_size) return;  // insufficient scratch

  auto gemm = [&](float* Y, _Float16* Yt, const _Float16* Xt,
                  const float* W, const float* bias, const float* add,
                  int Cin, int P, int ytStride, int ytOff,
                  long Xbs, long Ybs, long Ytbs, long Abs, int nb) {
    dim3 grid(P / 128, 1, nb);
    gemm128_wmma<<<grid, 256, 0, stream>>>(Y, Yt, Xt, W, bias, add, Cin, P,
                                           ytStride, ytOff, Xbs, Ybs, Ytbs, Abs);
  };
  auto transpose = [&](const float* X, _Float16* Yt, int C, int P,
                       int ytStride, int ytOff, long Xbs, long Ybs, int nb) {
    dim3 grid(P / 32, C / 32, nb);
    transpose_f16_kernel<<<grid, 256, 0, stream>>>(X, Yt, C, P, ytStride, ytOff, Xbs, Ybs);
  };
  auto gnorm = [&](float* X, const float* gam, const float* bet, int P) {
    zero_f32<<<1, 64, 0, stream>>>(gnstat, 2 * NGRP);
    gn_reduce<<<dim3(32, NGRP), 256, 0, stream>>>(X, gnstat, P);
    gn_apply_relu<<<cdivu((long)CDIM * P, 256), 256, 0, stream>>>(X, gnstat, gam, bet, P);
  };

  // ---------- phase A: sampling + geometry ----------
  zero_f32<<<cdivu(2L * BB * MM, 256), 256, 0, stream>>>(out2, 2L * BB * MM); // out2 & out3
  fps_kernel<<<BB, 1024, 0, stream>>>(xyzs, sidx);
  sample_xyz_kernel<<<cdivu((long)BB * MM, 256), 256, 0, stream>>>(xyzs, sidx, out0, spts);
  nearest_kernel<<<dim3(NN / 256, BB), 256, 0, stream>>>(spts, xyzs, o2s, out2);
  knn_kernel<<<dim3(MM, BB), 256, 0, stream>>>(spts, xyzs, knn);
  geom_kernel<<<cdivu((long)BB * MKc, 256), 256, 0, stream>>>(
      xyzs, spts, knn, o2s, negdiff, localf, maskb, out3);
  meandiv_kernel<<<cdivu((long)BB * MM, 256), 256, 0, stream>>>(out3, out2);

  // ---------- phase B: full-batch feature GEMMs ----------
  const long CN = (long)CDIM * NN, CM = (long)CDIM * MM;
  transpose(feats, featsT, CDIM, NN, CDIM, 0, CN, CN, BB);
  gemm(fpre, fpreT, featsT, prm[P_pre_w], prm[P_pre_b], nullptr,
       CDIM, NN, CDIM, 0, CN, CN, CN, 0, BB);
  gemm(wkf, nullptr, fpreT, prm[P_wk_w], prm[P_wk_b], nullptr,
       CDIM, NN, 0, 0, CN, CN, 0, 0, BB);
  gemm(wvf, nullptr, fpreT, prm[P_wv_w], prm[P_wv_b], nullptr,
       CDIM, NN, 0, 0, CN, CN, 0, 0, BB);
  gather_cm_kernel<<<cdivu((long)BB * CM, 256), 256, 0, stream>>>(sfeat, fpre, sidx);
  gather_cm_kernel<<<cdivu((long)BB * CM, 256), 256, 0, stream>>>(idg, feats, sidx);
  transpose(sfeat, sfeatT, CDIM, MM, CDIM, 0, CM, CM, BB);
  gemm(query, nullptr, sfeatT, prm[P_wq_w], prm[P_wq_b], nullptr,
       CDIM, MM, 0, 0, CM, CM, 0, 0, BB);

  // ---------- phase C: per-batch heavy pipeline ----------
  const unsigned gMK = cdivu((long)CDIM * MKc, 256);
  const unsigned gM  = cdivu((long)CDIM * MM, 256);
  for (int b = 0; b < BB; ++b) {
    const int*   knn_b  = knn + (long)b * MKc;
    const int*   mask_b = maskb + (long)b * MKc;
    const float* q_b    = query + (long)b * CM;
    const float* wkf_b  = wkf + (long)b * CN;
    const float* wvf_b  = wvf + (long)b * CN;
    const float* sf_b   = sfeat + (long)b * CM;
    const float* idg_b  = idg + (long)b * CM;
    float*       res_b  = resb + (long)b * CM;
    float*       out1_b = out1 + (long)b * CM;

    // pos_enc = mlp2(-diff, delta)  -> big0
    conv_small_cin<<<gMK, 256, 0, stream>>>(big1, negdiff + (long)b * 3 * MKc,
                                            prm[P_delta_w1], prm[P_delta_b1], 3, MKc);
    gnorm(big1, prm[P_delta_g], prm[P_delta_be], MKc);
    transpose(big1, bigT, CDIM, MKc, CDIM, 0, 0, 0, 1);
    gemm(big0, nullptr, bigT, prm[P_delta_w2], prm[P_delta_b2], nullptr,
         CDIM, MKc, 0, 0, 0, 0, 0, 0, 1);
    // t = query - key[knn] + pos_enc ; att = mlp2(t, gamma)/sqrt(HID)
    attn_in_kernel<<<gMK, 256, 0, stream>>>(big1, q_b, wkf_b, big0, knn_b);
    transpose(big1, bigT, CDIM, MKc, CDIM, 0, 0, 0, 1);
    gemm(big2, nullptr, bigT, prm[P_gamma_w1], prm[P_gamma_b1], nullptr,
         CDIM, MKc, 0, 0, 0, 0, 0, 0, 1);
    gnorm(big2, prm[P_gamma_g], prm[P_gamma_be], MKc);
    transpose(big2, bigT, CDIM, MKc, CDIM, 0, 0, 0, 1);
    gemm(big1, nullptr, bigT, prm[P_gamma_w2], prm[P_gamma_b2], nullptr,
         CDIM, MKc, 0, 0, 0, 0, 0, 0, 1);
    softmax16_kernel<<<gM, 256, 0, stream>>>(big1, mask_b, 0.08838834764831845f);
    attn_out_kernel<<<gM, 256, 0, stream>>>(res_b, big1, wvf_b, big0, knn_b);
    // ancestor = conv(res) + sampled_feats -> cat3 rows [0,128) (+ catT)
    transpose(res_b, resT, CDIM, MM, CDIM, 0, 0, 0, 1);
    gemm(cat3, catT, resT, prm[P_ptpost_w], prm[P_ptpost_b], sf_b,
         CDIM, MM, 3 * CDIM, 0, 0, 0, 0, 0, 1);

    // position embedding: pe = mlp2(local, pospre); pa = softmax(mlp2(pe, posattn))
    conv_small_cin<<<gMK, 256, 0, stream>>>(big1, localf + (long)b * 4 * MKc,
                                            prm[P_pospre_w1], prm[P_pospre_b1], 4, MKc);
    gnorm(big1, prm[P_pospre_g], prm[P_pospre_be], MKc);
    transpose(big1, bigT, CDIM, MKc, CDIM, 0, 0, 0, 1);
    gemm(big2, bigTb, bigT, prm[P_pospre_w2], prm[P_pospre_b2], nullptr,
         CDIM, MKc, CDIM, 0, 0, 0, 0, 0, 1);          // pe (f32 + f16T)
    gemm(big1, nullptr, bigTb, prm[P_posattn_w1], prm[P_posattn_b1], nullptr,
         CDIM, MKc, 0, 0, 0, 0, 0, 0, 1);
    gnorm(big1, prm[P_posattn_g], prm[P_posattn_be], MKc);
    transpose(big1, bigT, CDIM, MKc, CDIM, 0, 0, 0, 1);
    gemm(big0, nullptr, bigT, prm[P_posattn_w2], prm[P_posattn_b2], nullptr,
         CDIM, MKc, 0, 0, 0, 0, 0, 0, 1);             // pa
    softmax16_kernel<<<gM, 256, 0, stream>>>(big0, mask_b, 1.0f);
    mulsum16_kernel<<<gM, 256, 0, stream>>>(cat3 + (long)CDIM * MM, big2, big0);
    transpose(cat3 + (long)CDIM * MM, catT, CDIM, MM, 3 * CDIM, CDIM, 0, 0, 1);

    // density embedding -> cat3 rows [256,384) (+ catT)
    conv_small_cin<<<gM, 256, 0, stream>>>(denh, out2 + (long)b * MM,
                                           prm[P_den_w1], prm[P_den_b1], 1, MM);
    gnorm(denh, prm[P_den_g], prm[P_den_be], MM);
    transpose(denh, denhT, CDIM, MM, CDIM, 0, 0, 0, 1);
    gemm(cat3 + (long)2 * CDIM * MM, catT, denhT, prm[P_den_w2], prm[P_den_b2],
         nullptr, CDIM, MM, 3 * CDIM, 2 * CDIM, 0, 0, 0, 0, 1);

    // final projection + identity residual
    gemm(out1_b, nullptr, catT, prm[P_post_w], prm[P_post_b], idg_b,
         3 * CDIM, MM, 0, 0, 0, 0, 0, 0, 1);
  }
}
